// Attention_70059506532928
// MI455X (gfx1250) — matrix operand order repaired and checked by reading, not compiled, not run on hardware
//
#include <hip/hip_runtime.h>

// ---------------------------------------------------------------------------
// MI455X (gfx1250, wave32) attention layer:
//   f32->bf16 conversion pass -> QKV proj (bf16 WMMA, async-LDS double
//   buffered) -> RMSNorm+RoPE -> flash attention (bf16 WMMA, async K tiles)
//   -> output proj (bf16 WMMA). All accumulation in f32.
// ---------------------------------------------------------------------------

typedef __attribute__((ext_vector_type(16))) __bf16 v16bf;
typedef __attribute__((ext_vector_type(8)))  float  v8f;
typedef int v4i_vs __attribute__((vector_size(16)));   // matches builtin param

#define S_LEN  2048
#define DMODEL 2560
#define NHEADS 32
#define NKV    8
#define HDIM   128

// CDNA5 async global->LDS DMA (ASYNCcnt-tracked). Guarded: falls back to
// VGPR staging if this toolchain does not declare the builtins.
#if defined(__HIP_DEVICE_COMPILE__) && defined(__has_builtin)
#if __has_builtin(__builtin_amdgcn_global_load_async_to_lds_b128) && \
    __has_builtin(__builtin_amdgcn_s_wait_asynccnt)
#define USE_ASYNC_LDS 1
#endif
#endif
#ifndef USE_ASYNC_LDS
#define USE_ASYNC_LDS 0
#endif

__device__ __forceinline__ void cp16_g2l(const unsigned short* g, unsigned short* l) {
#if USE_ASYNC_LDS
  __builtin_amdgcn_global_load_async_to_lds_b128(
      (__attribute__((address_space(1))) v4i_vs*)(g),
      (__attribute__((address_space(3))) v4i_vs*)(l), 0, 0);
#else
  *(uint4*)l = *(const uint4*)g;
#endif
}
#if USE_ASYNC_LDS
#define ASYNC_WAIT0() __builtin_amdgcn_s_wait_asynccnt(0)
#else
#define ASYNC_WAIT0() ((void)0)
#endif

union Frag { uint4 q[2]; v16bf v; };
static_assert(sizeof(Frag) == 32, "frag size");

__device__ __forceinline__ unsigned short f2bf(float f) {
  union { float f; unsigned u; } c; c.f = f;
  unsigned u = c.u;
  u += 0x7FFFu + ((u >> 16) & 1u);          // round-to-nearest-even
  return (unsigned short)(u >> 16);
}
__device__ __forceinline__ unsigned pack2(float a, float b) {
  return (unsigned)f2bf(a) | ((unsigned)f2bf(b) << 16);
}

// ---------------------------------------------------------------------------
// Bulk f32 -> bf16 (vectorized, one float4 / uint2 per thread)
// ---------------------------------------------------------------------------
__global__ __launch_bounds__(256) void to_bf16(const float* __restrict__ src,
                                               unsigned short* __restrict__ dst,
                                               int n4) {
  const int i = blockIdx.x * 256 + threadIdx.x;
  if (i < n4) {
    const float4 v = ((const float4*)src)[i];
    ((uint2*)dst)[i] = make_uint2(pack2(v.x, v.y), pack2(v.z, v.w));
  }
}

// ---------------------------------------------------------------------------
// C[M,N] = A[M,K] * B[N,K]^T, A/B bf16 (K-contiguous rows), C f32, M = 2048.
// 128x128 C tile, 256 threads = 8 waves (4 over M x 2 over N),
// wave: 2x4 grid of 16x16x32 WMMA accumulators.
// Double-buffered LDS tiles; async global->LDS staging issued one tile ahead.
// ---------------------------------------------------------------------------
template <int N, int K>
__global__ __launch_bounds__(256) void gemm_bf16_t(
    const unsigned short* __restrict__ A,
    const unsigned short* __restrict__ B,
    float* __restrict__ C)
{
  __shared__ unsigned short lsa[2][128 * 40];   // A tile 128 x 32 (+8 pad)
  __shared__ unsigned short lsb[2][128 * 40];   // B tile 128 x 32 (+8 pad)

  const int tid  = threadIdx.x;
  const int lane = tid & 31;
  const int wid  = tid >> 5;
  const int wm   = wid & 3;
  const int wn   = wid >> 2;
  const int bm   = blockIdx.y * 128;
  const int bn   = blockIdx.x * 128;

  const int srow = (tid >> 2);                  // staging: 2 uint4 per thread
  const int sc8  = (tid & 3) * 8;

  auto stage = [&](int buf, int k0) {
#pragma unroll
    for (int t = 0; t < 2; ++t) {
      const int row = srow + t * 64;
      cp16_g2l(A + (bm + row) * K + k0 + sc8, &lsa[buf][row * 40 + sc8]);
      cp16_g2l(B + (bn + row) * K + k0 + sc8, &lsb[buf][row * 40 + sc8]);
    }
  };

  const v8f vzero = {0.f,0.f,0.f,0.f,0.f,0.f,0.f,0.f};
  v8f acc[2][4];
#pragma unroll
  for (int mt = 0; mt < 2; ++mt)
#pragma unroll
    for (int nt = 0; nt < 4; ++nt) acc[mt][nt] = vzero;

  const int rl  = lane & 15;
  const int ac0 = (lane < 16) ? 0 : 8;          // A-frag K-half select
  const int bk0 = (lane < 16) ? 0 : 8;          // B-frag K-half select

  constexpr int NK = K / 32;
  stage(0, 0);
  for (int kt = 0; kt < NK; ++kt) {
    const int cur = kt & 1;
    ASYNC_WAIT0();                               // this thread's cur tile landed
    __syncthreads();                             // everyone's tile landed / prev compute done
    if (kt + 1 < NK) stage(cur ^ 1, (kt + 1) * 32);

    Frag af[2], bf[4];
#pragma unroll
    for (int mt = 0; mt < 2; ++mt) {
      const int r = wm * 32 + mt * 16 + rl;
      af[mt].q[0] = *(const uint4*)&lsa[cur][r * 40 + ac0];
      af[mt].q[1] = *(const uint4*)&lsa[cur][r * 40 + ac0 + 16];
    }
#pragma unroll
    for (int nt = 0; nt < 4; ++nt) {
      const int n = wn * 64 + nt * 16 + rl;
      bf[nt].q[0] = *(const uint4*)&lsb[cur][n * 40 + bk0];
      bf[nt].q[1] = *(const uint4*)&lsb[cur][n * 40 + bk0 + 16];
    }
#pragma unroll
    for (int mt = 0; mt < 2; ++mt)
#pragma unroll
      for (int nt = 0; nt < 4; ++nt)
        acc[mt][nt] = __builtin_amdgcn_wmma_f32_16x16x32_bf16(
            false, af[mt].v, false, bf[nt].v, (short)0, acc[mt][nt], false, false);
  }

  const int mh = (lane >= 16) ? 8 : 0;
#pragma unroll
  for (int mt = 0; mt < 2; ++mt)
#pragma unroll
    for (int nt = 0; nt < 4; ++nt) {
      float* crow = C + (bm + wm * 32 + mt * 16 + mh) * N +
                    bn + wn * 64 + nt * 16 + rl;
#pragma unroll
      for (int r = 0; r < 8; ++r) crow[r * N] = acc[mt][nt][r];
    }
}

// ---------------------------------------------------------------------------
// RMSNorm (per head, dim 128) + RoPE. One wave per (s, head).
// ---------------------------------------------------------------------------
__global__ __launch_bounds__(32) void rmsnorm_rope_q(
    const float* __restrict__ q, const float* __restrict__ w,
    unsigned short* __restrict__ qhat)
{
  const int s = blockIdx.x, h = blockIdx.y, lane = threadIdx.x;
  const float* base = q + (size_t)s * (NHEADS * HDIM) + h * HDIM;
  const int d0 = lane * 4;
  const float4 xv = *(const float4*)(base + d0);
  float ss = xv.x * xv.x + xv.y * xv.y + xv.z * xv.z + xv.w * xv.w;
#pragma unroll
  for (int m = 1; m < 32; m <<= 1) ss += __shfl_xor(ss, m, 32);
  const float r = rsqrtf(ss * (1.f / HDIM) + 1e-6f);
  const int pd0 = (d0 < 64) ? d0 + 64 : d0 - 64;
  const float4 pv = *(const float4*)(base + pd0);
  const float xa[4] = {xv.x, xv.y, xv.z, xv.w};
  const float pa[4] = {pv.x, pv.y, pv.z, pv.w};
  const float pos = (float)s;
#pragma unroll
  for (int j = 0; j < 4; ++j) {
    const int d  = d0 + j;
    const float xn = xa[j] * r * w[d];
    const float pn = pa[j] * r * w[pd0 + j];
    const int idx = (d < 64) ? d : d - 64;
    const float ang = pos * powf(5000000.0f, -(float)(2 * idx) * (1.0f / 128.0f));
    const float cz = cosf(ang), sz = sinf(ang);
    const float o = (d < 64) ? (xn * cz - pn * sz) : (xn * cz + pn * sz);
    qhat[((size_t)h * S_LEN + s) * HDIM + d] = f2bf(o);
  }
}

__global__ __launch_bounds__(32) void rmsnorm_rope_k(
    const float* __restrict__ k, const float* __restrict__ w,
    float* __restrict__ kcache, unsigned short* __restrict__ khat)
{
  const int s = blockIdx.x, g = blockIdx.y, lane = threadIdx.x;
  const float* base = k + (size_t)s * (NKV * HDIM) + g * HDIM;
  const int d0 = lane * 4;
  const float4 xv = *(const float4*)(base + d0);
  float ss = xv.x * xv.x + xv.y * xv.y + xv.z * xv.z + xv.w * xv.w;
#pragma unroll
  for (int m = 1; m < 32; m <<= 1) ss += __shfl_xor(ss, m, 32);
  const float r = rsqrtf(ss * (1.f / HDIM) + 1e-6f);
  const int pd0 = (d0 < 64) ? d0 + 64 : d0 - 64;
  const float4 pv = *(const float4*)(base + pd0);
  const float xa[4] = {xv.x, xv.y, xv.z, xv.w};
  const float pa[4] = {pv.x, pv.y, pv.z, pv.w};
  const float pos = (float)s;
#pragma unroll
  for (int j = 0; j < 4; ++j) {
    const int d  = d0 + j;
    const float xn = xa[j] * r * w[d];
    const float pn = pa[j] * r * w[pd0 + j];
    const int idx = (d < 64) ? d : d - 64;
    const float ang = pos * powf(5000000.0f, -(float)(2 * idx) * (1.0f / 128.0f));
    const float cz = cosf(ang), sz = sinf(ang);
    const float o = (d < 64) ? (xn * cz - pn * sz) : (xn * cz + pn * sz);
    const size_t oi = ((size_t)g * S_LEN + s) * HDIM + d;
    kcache[oi] = o;                 // new_cache_k output (f32)
    khat[oi]   = f2bf(o);           // bf16 for attention
  }
}

// V: reorder [s][g*128+d] -> [g][s][d], emit f32 cache + bf16 for attention
__global__ __launch_bounds__(32) void v_reorder(
    const float* __restrict__ v, float* __restrict__ vcache,
    unsigned short* __restrict__ vhat)
{
  const int s = blockIdx.x, g = blockIdx.y, lane = threadIdx.x;
  const float4 xv =
      *(const float4*)(v + (size_t)s * (NKV * HDIM) + g * HDIM + lane * 4);
  const size_t o = ((size_t)g * S_LEN + s) * HDIM + lane * 4;
  *(float4*)(vcache + o) = xv;
  *(uint2*)&vhat[o] = make_uint2(pack2(xv.x, xv.y), pack2(xv.z, xv.w));
}

// ---------------------------------------------------------------------------
// Flash attention (causal, GQA 4:1). One block = 128 q-rows x 1 head.
// 8 waves; each wave owns 16 q-rows -> online softmax needs only
// intra-16-lane shuffles. Q fragments register-resident across the KV sweep.
// K tiles staged by async global->LDS DMA; V staged transposed via VGPRs.
// Output written directly in bf16 (consumer is the bf16 Wo GEMM).
// ---------------------------------------------------------------------------
__global__ __launch_bounds__(256) void flash_attn(
    const unsigned short* __restrict__ qhat,
    const unsigned short* __restrict__ khat,
    const unsigned short* __restrict__ vhat,
    unsigned short* __restrict__ o)
{
  __shared__ unsigned short kt[32 * 136];    // K tile 32(kv) x 128(d), pad 8
  __shared__ unsigned short vt[128 * 40];    // V^T tile 128(d) x 32(kv), pad 8
  __shared__ unsigned short ps[8][16 * 32];  // per-wave P scratch (C->A relayout)

  const int tid  = threadIdx.x;
  const int lane = tid & 31;
  const int wid  = tid >> 5;
  const int q0   = blockIdx.x * 128;
  const int h    = blockIdx.y;
  const int g    = h >> 2;
  const int qr   = q0 + wid * 16;
  const int rl   = lane & 15;
  const int mh   = (lane >= 16) ? 8 : 0;
  const int ac0  = (lane < 16) ? 0 : 8;
  const int bk0  = (lane < 16) ? 0 : 8;

  Frag qf[4];
  {
    const unsigned short* qrow = qhat + ((size_t)h * S_LEN + qr + rl) * HDIM;
#pragma unroll
    for (int c = 0; c < 4; ++c) {
      qf[c].q[0] = *(const uint4*)(qrow + c * 32 + ac0);
      qf[c].q[1] = *(const uint4*)(qrow + c * 32 + ac0 + 16);
    }
  }

  const v8f vzero = {0.f,0.f,0.f,0.f,0.f,0.f,0.f,0.f};
  v8f oacc[8];
#pragma unroll
  for (int i = 0; i < 8; ++i) oacc[i] = vzero;
  float mcur[8], lcur[8];
#pragma unroll
  for (int r = 0; r < 8; ++r) { mcur[r] = -1e30f; lcur[r] = 0.f; }

  const float scale = 0.088388347648318447f;  // 1/sqrt(128)
  const int jmax = (q0 + 127) >> 5;
  const int srow = tid >> 4;                  // staging: uint4 per thread x2
  const int sc8  = (tid & 15) * 8;

  for (int j = 0; j <= jmax; ++j) {
    const int kv0 = j * 32;
    __syncthreads();                           // previous tile fully consumed
    // stage K [32x128] via async DMA; V^T [128x32] via VGPR transpose
#pragma unroll
    for (int t = 0; t < 2; ++t) {
      const int row = srow + t * 16;
      cp16_g2l(khat + ((size_t)g * S_LEN + kv0 + row) * HDIM + sc8,
               &kt[row * 136 + sc8]);
      const uint4 v4 =
          *(const uint4*)(vhat + ((size_t)g * S_LEN + kv0 + row) * HDIM + sc8);
      union { uint4 q; unsigned short u[8]; } vu; vu.q = v4;
#pragma unroll
      for (int e = 0; e < 8; ++e) vt[(sc8 + e) * 40 + row] = vu.u[e];
    }
    ASYNC_WAIT0();
    __syncthreads();

    // S = Q K^T  (16 rows x 32 kv, accumulated over d in 4 WMMAs per subtile)
    v8f sacc[2]; sacc[0] = vzero; sacc[1] = vzero;
#pragma unroll
    for (int sub = 0; sub < 2; ++sub) {
#pragma unroll
      for (int c = 0; c < 4; ++c) {
        Frag bf;
        const int kr = sub * 16 + rl;
        bf.q[0] = *(const uint4*)&kt[kr * 136 + c * 32 + bk0];
        bf.q[1] = *(const uint4*)&kt[kr * 136 + c * 32 + bk0 + 16];
        sacc[sub] = __builtin_amdgcn_wmma_f32_16x16x32_bf16(
            false, qf[c].v, false, bf.v, (short)0, sacc[sub], false, false);
      }
    }

    // online softmax over the wave's 16 rows (rows (r, half) per lane)
    float corr[8];
#pragma unroll
    for (int r = 0; r < 8; ++r) {
      const int qrow = qr + r + mh;
      float s0 = sacc[0][r] * scale; if (kv0 + rl      > qrow) s0 = -1e30f;
      float s1 = sacc[1][r] * scale; if (kv0 + 16 + rl > qrow) s1 = -1e30f;
      float mx = fmaxf(s0, s1);
#pragma unroll
      for (int m2 = 1; m2 < 16; m2 <<= 1) mx = fmaxf(mx, __shfl_xor(mx, m2, 32));
      const float nm = fmaxf(mcur[r], mx);
      const float p0 = (s0 <= -1e29f) ? 0.f : expf(s0 - nm);
      const float p1 = (s1 <= -1e29f) ? 0.f : expf(s1 - nm);
      float rs = p0 + p1;
#pragma unroll
      for (int m2 = 1; m2 < 16; m2 <<= 1) rs += __shfl_xor(rs, m2, 32);
      const float cf = (mcur[r] <= -1e29f) ? 0.f : expf(mcur[r] - nm);
      lcur[r] = lcur[r] * cf + rs;
      mcur[r] = nm;
      corr[r] = cf;
      ps[wid][(r + mh) * 32 + rl]      = f2bf(p0);
      ps[wid][(r + mh) * 32 + 16 + rl] = f2bf(p1);
    }

    // C-layout -> A-layout via per-wave LDS (same-wave DS ops are in order)
    Frag pf;
    pf.q[0] = *(const uint4*)&ps[wid][rl * 32 + ac0];
    pf.q[1] = *(const uint4*)&ps[wid][rl * 32 + ac0 + 16];

    // O = diag(corr) * O + P * V
#pragma unroll
    for (int dt = 0; dt < 8; ++dt) {
#pragma unroll
      for (int r = 0; r < 8; ++r) oacc[dt][r] *= corr[r];
      Frag vf;
      vf.q[0] = *(const uint4*)&vt[(dt * 16 + rl) * 40 + bk0];
      vf.q[1] = *(const uint4*)&vt[(dt * 16 + rl) * 40 + bk0 + 16];
      oacc[dt] = __builtin_amdgcn_wmma_f32_16x16x32_bf16(
          false, pf.v, false, vf.v, (short)0, oacc[dt], false, false);
    }
  }

  // epilogue: divide by l, write bf16 [s][h*128+d] (K-major for the Wo GEMM)
#pragma unroll
  for (int dt = 0; dt < 8; ++dt)
#pragma unroll
    for (int r = 0; r < 8; ++r)
      o[(size_t)(qr + r + mh) * (NHEADS * HDIM) + h * HDIM + dt * 16 + rl] =
          f2bf(oacc[dt][r] / lcur[r]);
}

// ---------------------------------------------------------------------------
extern "C" void kernel_launch(void* const* d_in, const int* in_sizes, int n_in,
                              void* d_out, int out_size, void* d_ws, size_t ws_size,
                              hipStream_t stream) {
  (void)in_sizes; (void)n_in; (void)out_size; (void)ws_size;
  const float* x   = (const float*)d_in[0];
  const float* Wq  = (const float*)d_in[1];
  const float* Wk  = (const float*)d_in[2];
  const float* Wv  = (const float*)d_in[3];
  const float* Wo  = (const float*)d_in[4];
  const float* qnw = (const float*)d_in[5];
  const float* knw = (const float*)d_in[6];

  float* out = (float*)d_out;
  float* kcache = out + (size_t)S_LEN * DMODEL;                 // 5,242,880
  float* vcache = kcache + (size_t)NKV * S_LEN * HDIM;          // +2,097,152

  typedef unsigned short u16;
  char* ws = (char*)d_ws;
  u16*   xh  = (u16*)(ws);                       // [2048][2560] bf16  10.5 MB
  u16*   Wqh = (u16*)(ws + 10485760);            // [4096][2560] bf16  21.0 MB
  u16*   Wkh = (u16*)(ws + 31457280);            // [1024][2560] bf16   5.2 MB
  u16*   Wvh = (u16*)(ws + 36700160);            // [1024][2560] bf16   5.2 MB
  u16*   Woh = (u16*)(ws + 41943040);            // [2560][4096] bf16  21.0 MB
  float* qb  = (float*)(ws + 62914560);          // [2048][4096] f32   33.5 MB
  float* kb  = (float*)(ws + 96468992);          // [2048][1024] f32    8.4 MB
  float* vb  = (float*)(ws + 104857600);         // [2048][1024] f32    8.4 MB
  u16*   qh  = (u16*)(ws + 113246208);           // [32][2048][128] bf16
  u16*   kh  = (u16*)(ws + 130023424);           // [8][2048][128] bf16
  u16*   vh  = (u16*)(ws + 134217728);           // [8][2048][128] bf16
  u16*   oh  = (u16*)qb;                         // attn out bf16, reuses dead Q f32

  // one-shot f32 -> bf16 conversions (bandwidth pass, ~0.19 GB total)
  to_bf16<<<dim3(5120),  256, 0, stream>>>(x,  xh,  1310720);
  to_bf16<<<dim3(10240), 256, 0, stream>>>(Wq, Wqh, 2621440);
  to_bf16<<<dim3(2560),  256, 0, stream>>>(Wk, Wkh, 655360);
  to_bf16<<<dim3(2560),  256, 0, stream>>>(Wv, Wvh, 655360);
  to_bf16<<<dim3(10240), 256, 0, stream>>>(Wo, Woh, 2621440);

  // Q/K/V projections: C = x @ W^T
  gemm_bf16_t<NHEADS * HDIM, DMODEL><<<dim3(32, 16), 256, 0, stream>>>(xh, Wqh, qb);
  gemm_bf16_t<NKV * HDIM,    DMODEL><<<dim3( 8, 16), 256, 0, stream>>>(xh, Wkh, kb);
  gemm_bf16_t<NKV * HDIM,    DMODEL><<<dim3( 8, 16), 256, 0, stream>>>(xh, Wvh, vb);

  // RMSNorm + RoPE (K also emits the f32 cache output); V reorder
  rmsnorm_rope_q<<<dim3(S_LEN, NHEADS), 32, 0, stream>>>(qb, qnw, qh);
  rmsnorm_rope_k<<<dim3(S_LEN, NKV),    32, 0, stream>>>(kb, knw, kcache, kh);
  v_reorder     <<<dim3(S_LEN, NKV),    32, 0, stream>>>(vb, vcache, vh);

  // causal flash attention (writes bf16 attn output)
  flash_attn<<<dim3(S_LEN / 128, NHEADS), 256, 0, stream>>>(qh, kh, vh, oh);

  // output projection: out = attn_out @ Wo^T
  gemm_bf16_t<DMODEL, NHEADS * HDIM><<<dim3(20, 16), 256, 0, stream>>>(oh, Woh, out);
}
